// MSAPairWeightedAveraging_64424509440678
// MI455X (gfx1250) — matrix operand order repaired and checked by reading, not compiled, not run on hardware
//
#include <hip/hip_runtime.h>
#include <hip/hip_bf16.h>
#include <cstdint>
#include <cstddef>

// Problem constants
#define S_   256
#define N_   512
#define DM_  64
#define DP_  128
#define H_   8
#define DH_  32
#define DI_  256   // H*DH
#define DVG_ 512   // 2*DI

typedef __attribute__((ext_vector_type(16))) __bf16 v16bf;
typedef __attribute__((ext_vector_type(2)))  __bf16 v2bf;
typedef __attribute__((ext_vector_type(8)))  float  v8f;

union V16U { v16bf v; uint4 u[2]; };

// Native bf16 conversion (RNE) -> lets clang emit v_cvt_pk_bf16_f32
__device__ __forceinline__ unsigned short f2bf(float f) {
    __bf16 h = (__bf16)f;
    return __builtin_bit_cast(unsigned short, h);
}
__device__ __forceinline__ uint pack_bf16(float a, float b) {
    v2bf t;
    t[0] = (__bf16)a;
    t[1] = (__bf16)b;
    return __builtin_bit_cast(uint, t);
}
__device__ __forceinline__ float bfu(uint u, int half) {
    return __builtin_bit_cast(float, (half ? (u & 0xffff0000u) : (u << 16)));
}
__device__ __forceinline__ v8f vz8() { v8f z = {0.f,0.f,0.f,0.f,0.f,0.f,0.f,0.f}; return z; }
__device__ __forceinline__ float fast_sigmoid(float x) {
    return __builtin_amdgcn_rcpf(1.f + __expf(-x));
}

// ---------------------------------------------------------------------------
// K0: LayerNorm(msa) -> bf16 rows  [S*N, DM]
// ---------------------------------------------------------------------------
__global__ __launch_bounds__(256) void k0_ln_msa(
    const float* __restrict__ msa, const float* __restrict__ g,
    const float* __restrict__ b, unsigned short* __restrict__ xln)
{
    int row = blockIdx.x * 256 + threadIdx.x;          // 131072 rows
    const float* x = msa + (size_t)row * DM_;
    float s = 0.f, s2 = 0.f;
    #pragma unroll
    for (int k = 0; k < DM_; ++k) { float t = x[k]; s += t; s2 += t * t; }
    float m   = s  * (1.f / DM_);
    float var = s2 * (1.f / DM_) - m * m;
    float inv = __builtin_amdgcn_rsqf(var + 1e-5f);
    uint pk[DM_ / 2];
    #pragma unroll
    for (int k = 0; k < DM_; k += 2) {
        float y0 = (x[k]     - m) * inv * g[k]     + b[k];
        float y1 = (x[k + 1] - m) * inv * g[k + 1] + b[k + 1];
        pk[k / 2] = pack_bf16(y0, y1);
    }
    uint4* o4 = (uint4*)(xln + (size_t)row * DM_);
    #pragma unroll
    for (int q = 0; q < DM_ / 8; ++q)
        o4[q] = make_uint4(pk[4*q], pk[4*q+1], pk[4*q+2], pk[4*q+3]);
}

// ---------------------------------------------------------------------------
// K-setup: transpose + convert weights to bf16 (B-operand layout: K-contig)
// ---------------------------------------------------------------------------
__global__ __launch_bounds__(256) void k_setup_w(
    const float* __restrict__ w_vg, const float* __restrict__ w_out,
    unsigned short* __restrict__ wvgt, unsigned short* __restrict__ woutt)
{
    int idx = blockIdx.x * 256 + threadIdx.x;
    if (idx < DVG_ * DM_) {
        int n = idx / DM_, k = idx % DM_;
        wvgt[idx] = f2bf(w_vg[(size_t)k * DVG_ + n]);
    }
    if (idx < DM_ * DI_) {
        int dm = idx / DI_, k = idx % DI_;
        woutt[idx] = f2bf(w_out[(size_t)k * DM_ + dm]);
    }
}

// ---------------------------------------------------------------------------
// K1: LN(pair) @ w_b, mask, softmax over j -> bf16 weights [H, N, N]
// ---------------------------------------------------------------------------
__global__ __launch_bounds__(256) void k1_pair_softmax(
    const float* __restrict__ pair, const unsigned char* __restrict__ mask,
    const float* __restrict__ gp, const float* __restrict__ bp,
    const float* __restrict__ w_b, unsigned short* __restrict__ wattn)
{
    __shared__ float logits[H_][N_];
    __shared__ float wbs[DP_][H_];
    __shared__ float gs[DP_], bs[DP_];
    __shared__ float red[256];
    int tid = threadIdx.x, i = blockIdx.x;

    for (int k = tid; k < DP_ * H_; k += 256) wbs[k / H_][k % H_] = w_b[k];
    for (int k = tid; k < DP_; k += 256) { gs[k] = gp[k]; bs[k] = bp[k]; }
    __syncthreads();

    #pragma unroll
    for (int jj = 0; jj < 2; ++jj) {
        int j = tid + jj * 256;
        const float* x = pair + ((size_t)i * N_ + j) * DP_;
        float s = 0.f, s2 = 0.f;
        for (int k = 0; k < DP_; k += 4) {
            float4 t = *(const float4*)(x + k);
            s  += t.x + t.y + t.z + t.w;
            s2 += t.x*t.x + t.y*t.y + t.z*t.z + t.w*t.w;
        }
        float m = s * (1.f / DP_);
        float inv = __builtin_amdgcn_rsqf(s2 * (1.f / DP_) - m * m + 1e-5f);
        float acc[H_];
        #pragma unroll
        for (int h = 0; h < H_; ++h) acc[h] = 0.f;
        for (int k = 0; k < DP_; k += 4) {
            float4 t = *(const float4*)(x + k);
            float e[4] = { t.x, t.y, t.z, t.w };
            #pragma unroll
            for (int c = 0; c < 4; ++c) {
                float ln = (e[c] - m) * inv * gs[k + c] + bs[k + c];
                #pragma unroll
                for (int h = 0; h < H_; ++h) acc[h] += ln * wbs[k + c][h];
            }
        }
        bool mk = mask[j] != 0;
        #pragma unroll
        for (int h = 0; h < H_; ++h)
            logits[h][j] = mk ? acc[h] : -3.3895e38f;
    }
    __syncthreads();

    for (int h = 0; h < H_; ++h) {
        float lm = fmaxf(logits[h][tid], logits[h][tid + 256]);
        red[tid] = lm; __syncthreads();
        for (int off = 128; off > 0; off >>= 1) {
            if (tid < off) red[tid] = fmaxf(red[tid], red[tid + off]);
            __syncthreads();
        }
        float mx = red[0]; __syncthreads();
        float e0 = __expf(logits[h][tid] - mx);
        float e1 = __expf(logits[h][tid + 256] - mx);
        red[tid] = e0 + e1; __syncthreads();
        for (int off = 128; off > 0; off >>= 1) {
            if (tid < off) red[tid] += red[tid + off];
            __syncthreads();
        }
        float rs = __builtin_amdgcn_rcpf(red[0]); __syncthreads();
        size_t base = ((size_t)h * N_ + i) * N_;
        wattn[base + tid]       = f2bf(e0 * rs);
        wattn[base + tid + 256] = f2bf(e1 * rs);
    }
}

// ---------------------------------------------------------------------------
// K2: vg = Xln @ Wvg via WMMA bf16. One wave per 16-row tile.
//   values -> Vt[H][S][DH][N] bf16 (j-contiguous = einsum B-operand layout)
//   gates  -> Gt[H][S][DH][N] bf16 (same transposed layout, packed stores)
// ---------------------------------------------------------------------------
__global__ __launch_bounds__(128) void k2_vg_proj(
    const unsigned short* __restrict__ xln, const unsigned short* __restrict__ wvgt,
    unsigned short* __restrict__ vt, unsigned short* __restrict__ gts)
{
    int wave = threadIdx.x >> 5, lane = threadIdx.x & 31;
    int l16 = lane & 15, hi = lane >> 4;
    int tile = blockIdx.x * 4 + wave;                  // 8192 tiles
    int r0 = tile * 16;
    int sidx = r0 / N_;
    int nbase = r0 % N_;

    V16U a[2];                                         // A: Xln K-chunks (resident)
    const unsigned short* arow = xln + (size_t)(r0 + l16) * DM_;
    #pragma unroll
    for (int kc = 0; kc < 2; ++kc) {
        int klo = kc * 32 + hi * 16;
        a[kc].u[0] = *(const uint4*)(arow + klo);
        a[kc].u[1] = *(const uint4*)(arow + klo + 8);
    }

    auto loadB = [&](V16U* dst, int n0) {
        #pragma unroll
        for (int kc = 0; kc < 2; ++kc) {
            const unsigned short* bp = wvgt + (size_t)(n0 + l16) * DM_ + kc * 32 + hi * 16;
            dst[kc].u[0] = ((const uint4*)bp)[0];
            dst[kc].u[1] = ((const uint4*)bp)[1];
        }
    };

    V16U bc[2], bn[2];
    loadB(bc, 0);
    #pragma unroll 2
    for (int nt = 0; nt < 32; ++nt) {
        if (nt < 31) loadB(bn, (nt + 1) * 16);         // prefetch next tile
        v8f acc = vz8();
        acc = __builtin_amdgcn_wmma_f32_16x16x32_bf16(
            false, a[0].v, false, bc[0].v, (short)0, acc, false, false);
        acc = __builtin_amdgcn_wmma_f32_16x16x32_bf16(
            false, a[1].v, false, bc[1].v, (short)0, acc, false, false);
        if (nt < 16) {                                 // value tile -> Vt
            int h = nt >> 1;
            int d = (nt & 1) * 16 + l16;
            uint pk[4];
            #pragma unroll
            for (int r = 0; r < 4; ++r)
                pk[r] = pack_bf16(acc[2*r], acc[2*r+1]);
            size_t e = ((size_t)(h * S_ + sidx) * DH_ + d) * N_ + nbase + hi * 8;
            *(uint4*)(vt + e) = make_uint4(pk[0], pk[1], pk[2], pk[3]);
        } else {                                       // gate tile -> sigmoid -> Gt
            int t = nt - 16;
            int h = t >> 1;
            int d = (t & 1) * 16 + l16;
            uint pk[4];
            #pragma unroll
            for (int r = 0; r < 4; ++r)
                pk[r] = pack_bf16(fast_sigmoid(acc[2*r]), fast_sigmoid(acc[2*r+1]));
            size_t e = ((size_t)(h * S_ + sidx) * DH_ + d) * N_ + nbase + hi * 8;
            *(uint4*)(gts + e) = make_uint4(pk[0], pk[1], pk[2], pk[3]);
        }
        if (nt < 31) { bc[0] = bn[0]; bc[1] = bn[1]; }
    }
}

// ---------------------------------------------------------------------------
// K3: fused einsum + gating + out-projection (double-buffered K loop).
// ---------------------------------------------------------------------------
__global__ __launch_bounds__(128) void k3_attn_out(
    const unsigned short* __restrict__ wattn, const unsigned short* __restrict__ vt,
    const unsigned short* __restrict__ gts, const unsigned short* __restrict__ woutt,
    float* __restrict__ out)
{
    __shared__ float ctx[4][16][DI_];                  // 64 KB
    int wave = threadIdx.x >> 5, lane = threadIdx.x & 31;
    int l16 = lane & 15, hi = lane >> 4;
    int tile = blockIdx.x * 4 + wave;                  // 8192 tiles
    int s = tile >> 5;
    int i0 = (tile & 31) * 16;
    float (*cx)[DI_] = ctx[wave];

    for (int h = 0; h < H_; ++h) {
        v8f acc0 = vz8(), acc1 = vz8();
        const unsigned short* wrow  = wattn + ((size_t)h * N_ + i0 + l16) * N_;
        const unsigned short* vrow0 = vt + ((size_t)(h * S_ + s) * DH_ + l16) * N_;
        const unsigned short* vrow1 = vrow0 + (size_t)16 * N_;

        V16U am, b0, b1, amn, b0n, b1n;
        {
            int klo = hi * 16;
            am.u[0] = *(const uint4*)(wrow + klo);
            am.u[1] = *(const uint4*)(wrow + klo + 8);
            const unsigned short* bp0 = vrow0 + hi * 16;
            b0.u[0] = ((const uint4*)bp0)[0]; b0.u[1] = ((const uint4*)bp0)[1];
            const unsigned short* bp1 = vrow1 + hi * 16;
            b1.u[0] = ((const uint4*)bp1)[0]; b1.u[1] = ((const uint4*)bp1)[1];
        }
        #pragma unroll 2
        for (int jc = 0; jc < 16; ++jc) {
            if (jc < 15) {                             // prefetch next K-chunk
                int jn = (jc + 1) * 32;
                int klo = jn + hi * 16;
                amn.u[0] = *(const uint4*)(wrow + klo);
                amn.u[1] = *(const uint4*)(wrow + klo + 8);
                const unsigned short* bp0 = vrow0 + jn + hi * 16;
                b0n.u[0] = ((const uint4*)bp0)[0]; b0n.u[1] = ((const uint4*)bp0)[1];
                const unsigned short* bp1 = vrow1 + jn + hi * 16;
                b1n.u[0] = ((const uint4*)bp1)[0]; b1n.u[1] = ((const uint4*)bp1)[1];
            }
            acc0 = __builtin_amdgcn_wmma_f32_16x16x32_bf16(
                false, am.v, false, b0.v, (short)0, acc0, false, false);
            acc1 = __builtin_amdgcn_wmma_f32_16x16x32_bf16(
                false, am.v, false, b1.v, (short)0, acc1, false, false);
            if (jc < 15) { am = amn; b0 = b0n; b1 = b1n; }
        }
        // gates: packed loads from transposed Gt layout (8 consecutive rows/lane)
        size_t gb0 = ((size_t)(h * S_ + s) * DH_ + l16) * N_ + i0 + hi * 8;
        uint4 gp0 = *(const uint4*)(gts + gb0);
        uint4 gp1 = *(const uint4*)(gts + gb0 + (size_t)16 * N_);
        uint g0w[4] = { gp0.x, gp0.y, gp0.z, gp0.w };
        uint g1w[4] = { gp1.x, gp1.y, gp1.z, gp1.w };
        #pragma unroll
        for (int r = 0; r < 8; ++r) {
            int M = r + hi * 8;
            float g0 = bfu(g0w[r >> 1], r & 1);
            float g1 = bfu(g1w[r >> 1], r & 1);
            cx[M][h * DH_ + l16]      = acc0[r] * g0;
            cx[M][h * DH_ + 16 + l16] = acc1[r] * g1;
        }
    }
    __syncthreads();

    v8f accO[4] = { vz8(), vz8(), vz8(), vz8() };
    #pragma unroll 1
    for (int kc = 0; kc < 8; ++kc) {
        V16U am;
        int klo = kc * 32 + hi * 16;
        const float* crow = &cx[l16][klo];
        uint pk[8];
        #pragma unroll
        for (int t = 0; t < 4; ++t)
            pk[t]     = pack_bf16(crow[2*t],      crow[2*t+1]);
        #pragma unroll
        for (int t = 0; t < 4; ++t)
            pk[4 + t] = pack_bf16(crow[8 + 2*t], crow[8 + 2*t+1]);
        am.u[0] = make_uint4(pk[0], pk[1], pk[2], pk[3]);
        am.u[1] = make_uint4(pk[4], pk[5], pk[6], pk[7]);
        #pragma unroll
        for (int nt = 0; nt < 4; ++nt) {
            V16U bm;
            const unsigned short* bp = woutt + (size_t)(nt * 16 + l16) * DI_ + kc * 32 + hi * 16;
            bm.u[0] = ((const uint4*)bp)[0];
            bm.u[1] = ((const uint4*)bp)[1];
            accO[nt] = __builtin_amdgcn_wmma_f32_16x16x32_bf16(
                false, am.v, false, bm.v, (short)0, accO[nt], false, false);
        }
    }
    #pragma unroll
    for (int nt = 0; nt < 4; ++nt) {
        #pragma unroll
        for (int r = 0; r < 8; ++r) {
            int M = r + hi * 8;
            out[((size_t)s * N_ + i0 + M) * DM_ + nt * 16 + l16] = accO[nt][r];
        }
    }
}

// ---------------------------------------------------------------------------
extern "C" void kernel_launch(void* const* d_in, const int* in_sizes, int n_in,
                              void* d_out, int out_size, void* d_ws, size_t ws_size,
                              hipStream_t stream) {
    const float*         msa   = (const float*)d_in[0];
    const float*         pair  = (const float*)d_in[1];
    const unsigned char* mask  = (const unsigned char*)d_in[2];
    const float*         ln_mg = (const float*)d_in[3];
    const float*         ln_mb = (const float*)d_in[4];
    const float*         w_vg  = (const float*)d_in[5];
    const float*         ln_pg = (const float*)d_in[6];
    const float*         ln_pb = (const float*)d_in[7];
    const float*         w_b   = (const float*)d_in[8];
    const float*         w_out = (const float*)d_in[9];
    float* out = (float*)d_out;

    char* ws = (char*)d_ws;
    size_t oXLN   = 0;
    size_t oWVGT  = oXLN   + (size_t)S_ * N_ * DM_ * 2;        // 16.8 MB
    size_t oWOUTT = oWVGT  + (size_t)DVG_ * DM_ * 2;
    size_t oWATT  = oWOUTT + (size_t)DM_ * DI_ * 2;
    size_t oVT    = oWATT  + (size_t)H_ * N_ * N_ * 2;         // 4 MB
    size_t oGATES = oVT    + (size_t)H_ * S_ * DH_ * N_ * 2;   // 67 MB
    // end = oGATES + H*S*DH*N*2 = ~148 MB total

    unsigned short* xln   = (unsigned short*)(ws + oXLN);
    unsigned short* wvgt  = (unsigned short*)(ws + oWVGT);
    unsigned short* woutt = (unsigned short*)(ws + oWOUTT);
    unsigned short* watt  = (unsigned short*)(ws + oWATT);
    unsigned short* vt    = (unsigned short*)(ws + oVT);
    unsigned short* gts   = (unsigned short*)(ws + oGATES);

    k0_ln_msa<<<(S_ * N_) / 256, 256, 0, stream>>>(msa, ln_mg, ln_mb, xln);
    k_setup_w<<<(DVG_ * DM_) / 256, 256, 0, stream>>>(w_vg, w_out, wvgt, woutt);
    k1_pair_softmax<<<N_, 256, 0, stream>>>(pair, mask, ln_pg, ln_pb, w_b, watt);
    k2_vg_proj<<<(S_ * N_ / 16) / 4, 128, 0, stream>>>(xln, wvgt, vt, gts);
    k3_attn_out<<<(S_ * (N_ / 16)) / 4, 128, 0, stream>>>(watt, vt, gts, woutt, out);
}
